// RGN_21560735826305
// MI455X (gfx1250) — compile-verified, matrix operand order checked by reading, not run on hardware
//
#include <hip/hip_runtime.h>
#include <hip/hip_bf16.h>
#include <math.h>

// ---------------- problem constants ----------------
#define L_    1024
#define BT    64
#define ED_   512
#define HID   512
#define DIN   533           // ED + 21
#define KPAD  544           // 17*32, zero padded K for WMMA
#define G4H   2048          // 4*H gate rows
#define LU_   60
#define LUP   64            // padded head width
#define KLIN  1024          // 2*H
#define NBLK_DIR 16         // persistent blocks per LSTM direction

typedef __attribute__((ext_vector_type(16))) __bf16 v16bf;
typedef __attribute__((ext_vector_type(8)))  float  v8f;
typedef __attribute__((ext_vector_type(4)))  unsigned v4u;
typedef __attribute__((ext_vector_type(8)))  int    v8i;
typedef __attribute__((ext_vector_type(4)))  int    v4i;

#if defined(__has_builtin)
#  if __has_builtin(__builtin_amdgcn_tensor_load_to_lds)
#    define HAVE_TDM 1
#  endif
#endif

static __device__ __forceinline__ v8f wmma_bf16(v16bf a, v16bf b, v8f c) {
  return __builtin_amdgcn_wmma_f32_16x16x32_bf16(false, a, false, b, (short)0, c, false, false);
}
static __device__ __forceinline__ float sigmoidf_(float x) {
  return 1.0f / (1.0f + __expf(-x));
}
// CDNA5 async global->LDS copy (ASYNCcnt path)
static __device__ __forceinline__ void async_lds_b128(unsigned lds_byte_off, const void* g) {
  asm volatile("global_load_async_to_lds_b128 %0, %1, off"
               :: "v"(lds_byte_off), "v"(g) : "memory");
}
static __device__ __forceinline__ void wait_async0() {
  asm volatile("s_wait_asynccnt 0" ::: "memory");
}

// Tensor Data Mover: one contiguous 2D tile (nbytes8 x 1 rows of 8B elems) -> LDS
#ifdef HAVE_TDM
static __device__ __forceinline__ void tdm_load_contig(unsigned lds_off, const void* g,
                                                       unsigned n8 /* # of 8-byte elems */) {
  unsigned long long ga = (unsigned long long)(uintptr_t)g;
  v4u g0;
  g0[0] = 1u;                                          // count=1, user descriptor
  g0[1] = lds_off;                                     // lds_addr (bytes)
  g0[2] = (unsigned)(ga & 0xFFFFFFFFull);              // global_addr[31:0]
  g0[3] = (unsigned)((ga >> 32) & 0x1FFFFFFull) | (2u << 30); // addr[56:32] | type=2
  v8i g1;
  g1[0] = (int)(3u << 16);                             // data_size=8B, mask=0
  g1[1] = (int)((n8 & 0xFFFFu) << 16);                 // tensor_dim0[15:0]
  g1[2] = (int)(((n8 >> 16) & 0xFFFFu) | (1u << 16));  // tensor_dim0[31:16] | tensor_dim1=1
  g1[3] = (int)((n8 & 0xFFFFu) << 16);                 // tile_dim0 = n8 (<=65535)
  g1[4] = 1;                                           // tile_dim1 = 1
  g1[5] = (int)n8;                                     // tensor_dim0_stride[31:0]
  g1[6] = (int)((n8 & 0xFFFFu) << 16);                 // tensor_dim1_stride[15:0]
  g1[7] = 0;
  v4i z4 = {0, 0, 0, 0};
#if __clang_major__ >= 23
  v8i z8 = {0, 0, 0, 0, 0, 0, 0, 0};
  __builtin_amdgcn_tensor_load_to_lds(g0, g1, z4, z4, z8, 0);
#else
  __builtin_amdgcn_tensor_load_to_lds(g0, g1, z4, z4, 0);
#endif
}
#endif

// ---------------- workspace layout (bytes) ----------------
constexpr size_t OFF_XB   = 0;                                   // L*BT*KPAD bf16
constexpr size_t SZ_XB    = (size_t)L_ * BT * KPAD * 2;
constexpr size_t OFF_WIH  = OFF_XB + SZ_XB;                      // 2*G4H*KPAD bf16 (gate-permuted)
constexpr size_t SZ_WIH   = (size_t)2 * G4H * KPAD * 2;
constexpr size_t OFF_WHH  = OFF_WIH + SZ_WIH;                    // 2*G4H*HID bf16 (gate-permuted)
constexpr size_t SZ_WHH   = (size_t)2 * G4H * HID * 2;
constexpr size_t OFF_BIAS = OFF_WHH + SZ_WHH;                    // 2*G4H f32 (bih+bhh, permuted)
constexpr size_t SZ_BIAS  = (size_t)2 * G4H * 4;
constexpr size_t OFF_WLIN = OFF_BIAS + SZ_BIAS;                  // LUP*KLIN bf16
constexpr size_t SZ_WLIN  = (size_t)LUP * KLIN * 2;
constexpr size_t OFF_TRIG = OFF_WLIN + SZ_WLIN;                  // sin(180) + cos(180) f32
constexpr size_t SZ_TRIG  = 2048;
constexpr size_t OFF_SYNC = OFF_TRIG + SZ_TRIG;                  // per-dir {count,gen}
constexpr size_t SZ_SYNC  = 256;
constexpr size_t OFF_HBUF = OFF_SYNC + SZ_SYNC;                  // 2 dir * 2 parity * BT*HID bf16
constexpr size_t SZ_HBUF  = (size_t)2 * 2 * BT * HID * 2;
constexpr size_t OFF_XG   = OFF_HBUF + SZ_HBUF;                  // 2*L*BT*G4H f32
constexpr size_t SZ_XG    = (size_t)2 * L_ * BT * G4H * 4;
constexpr size_t OFF_HCAT = OFF_XG + SZ_XG;                      // L*BT*KLIN bf16 (concat h_f,h_b)
constexpr size_t SZ_HCAT  = (size_t)L_ * BT * KLIN * 2;
constexpr size_t OFF_LOG  = OFF_HCAT + SZ_HCAT;                  // L*BT*LUP f32
constexpr size_t SZ_LOG   = (size_t)L_ * BT * LUP * 4;
constexpr size_t OFF_PHI  = OFF_LOG + SZ_LOG;                    // L*BT*3 f32
constexpr size_t SZ_PHI   = (size_t)L_ * BT * 3 * 4;

// K3 LDS budget: Whh 128KB + z 32KB + h 64KB = 224KB (<= 320KB WGP LDS)
constexpr unsigned LDS_W_OFF = 0;
constexpr unsigned LDS_Z_OFF = 128u * 1024u;
constexpr unsigned LDS_H_OFF = 160u * 1024u;
constexpr unsigned LDS_TOTAL = 224u * 1024u;

// ---------------- K0a: pack Wih (gate-permuted, K padded, bf16) ----------------
__global__ void k_pack_wih(const float* wf, const float* wb, __bf16* out) {
  size_t idx = (size_t)blockIdx.x * blockDim.x + threadIdx.x;
  const size_t N = (size_t)2 * G4H * KPAD;
  if (idx >= N) return;
  int d  = (int)(idx / ((size_t)G4H * KPAD));
  size_t rem = idx % ((size_t)G4H * KPAD);
  int pr = (int)(rem / KPAD), k = (int)(rem % KPAD);
  int u = pr >> 2, j = pr & 3;                 // unit-major: pr = u*4 + gate
  int orig = j * HID + u;                      // torch gate order i,f,g,o
  const float* W = d ? wb : wf;
  float v = (k < DIN) ? W[(size_t)orig * DIN + k] : 0.0f;
  out[idx] = (__bf16)v;
}

// ---------------- K0b: pack Whh (gate-permuted, bf16) ----------------
__global__ void k_pack_whh(const float* wf, const float* wb, __bf16* out) {
  size_t idx = (size_t)blockIdx.x * blockDim.x + threadIdx.x;
  const size_t N = (size_t)2 * G4H * HID;
  if (idx >= N) return;
  int d  = (int)(idx / ((size_t)G4H * HID));
  size_t rem = idx % ((size_t)G4H * HID);
  int pr = (int)(rem / HID), k = (int)(rem % HID);
  int u = pr >> 2, j = pr & 3;
  int orig = j * HID + u;
  const float* W = d ? wb : wf;
  out[idx] = (__bf16)W[(size_t)orig * HID + k];
}

// ---------------- K0c: bias/wlin/trig/state-init ----------------
__global__ void k_pack_misc(const float* bihf, const float* bhhf,
                            const float* bihb, const float* bhhb,
                            const float* wlin, const float* alpha,
                            float* biasp, __bf16* wlinp,
                            float* sinT, float* cosT,
                            __bf16* hbuf, unsigned* syncc) {
  int idx = blockIdx.x * blockDim.x + threadIdx.x;
  if (idx < 4096) {                                   // permuted combined bias
    int d = idx >> 11, pr = idx & 2047;
    int u = pr >> 2, j = pr & 3, orig = j * HID + u;
    biasp[idx] = (d ? bihb[orig] : bihf[orig]) + (d ? bhhb[orig] : bhhf[orig]);
  } else if (idx < 4096 + LUP * KLIN) {               // W_lin padded 60->64 rows
    int r = idx - 4096;
    int n = r >> 10, k = r & 1023;
    wlinp[r] = (n < LU_) ? (__bf16)wlin[(size_t)n * KLIN + k] : (__bf16)0.0f;
  } else if (idx < 4096 + LUP * KLIN + 180) {         // alphabet sin/cos tables
    int r = idx - (4096 + LUP * KLIN);
    sinT[r] = sinf(alpha[r]);
    cosT[r] = cosf(alpha[r]);
  } else if (idx < 4096 + LUP * KLIN + 180 + (int)(2 * 2 * BT * HID)) {
    int r = idx - (4096 + LUP * KLIN + 180);
    hbuf[r] = (__bf16)0.0f;                           // zero h(-1) buffers
  } else if (idx < 4096 + LUP * KLIN + 180 + (int)(2 * 2 * BT * HID) + 4) {
    syncc[idx - (4096 + LUP * KLIN + 180 + (int)(2 * 2 * BT * HID))] = 0u;
  }
}

// ---------------- K1: build x = [embed[seq], pssm] as padded bf16 ----------------
__global__ void k_build_x(const int* seq, const float* pssm, const float* embed, __bf16* xb) {
  size_t idx = (size_t)blockIdx.x * blockDim.x + threadIdx.x;
  const size_t N = (size_t)L_ * BT * KPAD;
  if (idx >= N) return;
  size_t row = idx / KPAD;                            // l*BT + b
  int k = (int)(idx % KPAD);
  float v;
  if (k < ED_)       v = embed[(size_t)seq[row] * ED_ + k];
  else if (k < DIN)  v = pssm[row * 21 + (k - ED_)];
  else               v = 0.0f;
  xb[idx] = (__bf16)v;
}

// ---------------- K2: xg = x @ Wih^T + b  (both dirs), bf16 WMMA ----------------
__global__ void __launch_bounds__(256) k_gemm_xg(const __bf16* __restrict__ xb,
                                                 const __bf16* __restrict__ wih,
                                                 const float* __restrict__ biasp,
                                                 float* __restrict__ xg) {
  const int d  = blockIdx.z;
  const int m0 = blockIdx.x * 128;
  const int n0 = blockIdx.y * 128;
  const int tid = threadIdx.x, w = tid >> 5, lane = tid & 31;
  const int msub = (w & 1) * 64, nsub = (w >> 1) * 32;
  const int lhi = lane >> 4, llo = lane & 15;
  v8f acc[4][2];
  #pragma unroll
  for (int nt = 0; nt < 2; ++nt) {
    float bv = biasp[d * G4H + n0 + nsub + nt * 16 + llo];
    #pragma unroll
    for (int mt = 0; mt < 4; ++mt)
      #pragma unroll
      for (int i = 0; i < 8; ++i) acc[mt][nt][i] = bv;
  }
  #pragma unroll 1
  for (int kk = 0; kk < KPAD / 32; ++kk) {
    const int kb = kk * 32 + lhi * 16;
    v16bf bf[2];
    #pragma unroll
    for (int nt = 0; nt < 2; ++nt)
      bf[nt] = *(const v16bf*)(wih + ((size_t)d * G4H + n0 + nsub + nt * 16 + llo) * KPAD + kb);
    #pragma unroll
    for (int mt = 0; mt < 4; ++mt) {
      v16bf af = *(const v16bf*)(xb + (size_t)(m0 + msub + mt * 16 + llo) * KPAD + kb);
      #pragma unroll
      for (int nt = 0; nt < 2; ++nt) acc[mt][nt] = wmma_bf16(af, bf[nt], acc[mt][nt]);
    }
  }
  #pragma unroll
  for (int mt = 0; mt < 4; ++mt)
    #pragma unroll
    for (int nt = 0; nt < 2; ++nt)
      #pragma unroll
      for (int r = 0; r < 8; ++r) {
        int m = m0 + msub + mt * 16 + r + lhi * 8;
        int n = n0 + nsub + nt * 16 + llo;
        xg[((size_t)d * L_ * BT + m) * G4H + n] = acc[mt][nt][r];
      }
}

// ---------------- grid barrier (per direction, 16 blocks) ----------------
static __device__ __forceinline__ void grid_sync(unsigned* cnt, unsigned* gen) {
  __syncthreads();
  if (threadIdx.x == 0) {
    unsigned g = atomicAdd(gen, 0u);
    __threadfence();
    unsigned p = atomicAdd(cnt, 1u);
    if (p == NBLK_DIR - 1) {
      atomicExch(cnt, 0u);
      __threadfence();
      atomicAdd(gen, 1u);
    } else {
      while (atomicAdd(gen, 0u) == g) __builtin_amdgcn_s_sleep(1);
    }
  }
  __syncthreads();
}

// ---------------- K3: persistent biLSTM recurrence ----------------
// 32 blocks (16/dir), block owns 128 gate-rows == 32 hidden units.
// Whh slab (128KB) loaded to LDS once via TDM (TENSORcnt), h(t-1) staged to LDS
// each step via async-LDS (ASYNCcnt). Cell state c lives in registers.
__global__ void __launch_bounds__(256, 1) k_lstm(const float* __restrict__ xg,
                                                 const __bf16* __restrict__ whh,
                                                 __bf16* __restrict__ hbuf,
                                                 __bf16* __restrict__ hcat,
                                                 unsigned* __restrict__ syncc) {
  extern __shared__ __align__(16) char smem[];
  __bf16* sW = (__bf16*)(smem + LDS_W_OFF);   // 128*512 bf16 = 128 KB
  float*  sZ = (float*)(smem + LDS_Z_OFF);    // 64*128 f32  =  32 KB
  __bf16* sH = (__bf16*)(smem + LDS_H_OFF);   // 64*512 bf16 =  64 KB
  const int bx = blockIdx.x;
  const int d  = bx >> 4;
  const int nb = bx & 15;
  const int n0 = nb * 128;
  const int tid = threadIdx.x, w = tid >> 5, lane = tid & 31;
  const int lhi = lane >> 4, llo = lane & 15;

  // one-time DMA of the block's Whh slab (contiguous 128 KB) into LDS
  const char* wsrc = (const char*)(whh + ((size_t)d * G4H + n0) * HID);
#ifdef HAVE_TDM
  if (w == 0) {                                       // single wave drives the TDM
    tdm_load_contig(LDS_W_OFF, wsrc, (128u * HID * 2u) / 8u); // 16384 x 8B
    __builtin_amdgcn_s_wait_tensorcnt(0);
  }
#else
  #pragma unroll 1
  for (int it = 0; it < 32; ++it) {
    unsigned off = (unsigned)(it * 4096 + tid * 16);
    async_lds_b128(LDS_W_OFF + off, wsrc + off);
  }
  wait_async0();
#endif
  __syncthreads();

  __bf16* bufs = hbuf + (size_t)d * 2 * BT * HID;
  unsigned* cnt = syncc + d * 2;
  unsigned* gen = cnt + 1;

  float creg[8];
  #pragma unroll
  for (int e = 0; e < 8; ++e) creg[e] = 0.0f;

  #pragma unroll 1
  for (int s = 0; s < L_; ++s) {
    const int t = (d == 0) ? s : (L_ - 1 - s);
    const __bf16* hin  = bufs + (size_t)((s & 1) ^ 1) * BT * HID;
    __bf16*       hout = bufs + (size_t)(s & 1) * BT * HID;

    // stage h(t-1) (64 KB, read 8x by the 8 waves) into LDS asynchronously
    {
      const char* hsrc = (const char*)hin;
      #pragma unroll 1
      for (int it = 0; it < 16; ++it) {
        unsigned off = (unsigned)(it * 4096 + tid * 16);
        async_lds_b128(LDS_H_OFF + off, hsrc + off);
      }
      wait_async0();
      __syncthreads();
    }

    // z = h(t-1) @ Whh^T : 64x128, K=512; all operands from LDS
    v8f acc[4] = {};
    #pragma unroll 1
    for (int kk = 0; kk < HID / 32; ++kk) {
      const int kb = kk * 32 + lhi * 16;
      v16bf bf = *(const v16bf*)(sW + (size_t)(w * 16 + llo) * HID + kb);
      #pragma unroll
      for (int mt = 0; mt < 4; ++mt) {
        v16bf af = *(const v16bf*)(sH + (size_t)(mt * 16 + llo) * HID + kb);
        acc[mt] = wmma_bf16(af, bf, acc[mt]);
      }
    }
    #pragma unroll
    for (int mt = 0; mt < 4; ++mt)
      #pragma unroll
      for (int r = 0; r < 8; ++r)
        sZ[(mt * 16 + r + lhi * 8) * 128 + w * 16 + llo] = acc[mt][r];
    __syncthreads();

    // gates: 64 batch x 32 units = 2048 pairs; 8 per thread; xg read as float4
    const float* xgrow = xg + ((size_t)d * L_ + t) * BT * G4H;
    #pragma unroll
    for (int e = 0; e < 8; ++e) {
      int pair = e * 256 + tid;
      int ul = pair & 31, m = pair >> 5;
      const float4 x4 = *(const float4*)(xgrow + (size_t)m * G4H + n0 + ul * 4);
      float zi = x4.x + sZ[m * 128 + ul * 4 + 0];
      float zf = x4.y + sZ[m * 128 + ul * 4 + 1];
      float zg = x4.z + sZ[m * 128 + ul * 4 + 2];
      float zo = x4.w + sZ[m * 128 + ul * 4 + 3];
      float ig = sigmoidf_(zi), fg = sigmoidf_(zf);
      float gg = tanhf(zg),     og = sigmoidf_(zo);
      float c = fg * creg[e] + ig * gg;
      creg[e] = c;
      float h = og * tanhf(c);
      int u = (n0 >> 2) + ul;                         // global hidden unit
      hout[(size_t)m * HID + u] = (__bf16)h;
      hcat[((size_t)t * BT + m) * KLIN + d * HID + u] = (__bf16)h;
    }

    // prefetch next step's xg slab (global_prefetch_b8)
    if (s + 1 < L_) {
      const int tn = (d == 0) ? (s + 1) : (L_ - 2 - s);
      const float* nx = xg + ((size_t)d * L_ + tn) * BT * G4H + n0;
      __builtin_prefetch(nx + (size_t)(tid >> 2) * G4H + (tid & 3) * 32, 0, 1);
    }
    __threadfence();
    grid_sync(cnt, gen);
  }
}

// ---------------- K4: logits = hcat @ Wlin^T  (M=65536, K=1024, N=64) ----------------
__global__ void __launch_bounds__(256) k_gemm_logits(const __bf16* __restrict__ hcat,
                                                     const __bf16* __restrict__ wlinp,
                                                     float* __restrict__ logits) {
  const int m0 = blockIdx.x * 128;
  const int tid = threadIdx.x, w = tid >> 5, lane = tid & 31;
  const int lhi = lane >> 4, llo = lane & 15;
  v8f acc[4] = {};
  #pragma unroll 1
  for (int kk = 0; kk < KLIN / 32; ++kk) {
    const int kb = kk * 32 + lhi * 16;
    v16bf af = *(const v16bf*)(hcat + (size_t)(m0 + w * 16 + llo) * KLIN + kb);
    #pragma unroll
    for (int nt = 0; nt < 4; ++nt) {
      v16bf bf = *(const v16bf*)(wlinp + (size_t)(nt * 16 + llo) * KLIN + kb);
      acc[nt] = wmma_bf16(af, bf, acc[nt]);
    }
  }
  #pragma unroll
  for (int nt = 0; nt < 4; ++nt)
    #pragma unroll
    for (int r = 0; r < 8; ++r)
      logits[(size_t)(m0 + w * 16 + r + lhi * 8) * LUP + nt * 16 + llo] = acc[nt][r];
}

// ---------------- K5: angles (softmax normalizer cancels inside atan2) ----------------
__global__ void k_angles(const float* __restrict__ logits, const float* __restrict__ blin,
                         const float* __restrict__ sinT, const float* __restrict__ cosT,
                         float* __restrict__ phi) {
  int lb = blockIdx.x * blockDim.x + threadIdx.x;
  if (lb >= L_ * BT) return;
  const float* zr = logits + (size_t)lb * LUP;
  float zmax = -1e30f;
  for (int u = 0; u < LU_; ++u) {
    float z = zr[u] + blin[u];
    zmax = fmaxf(zmax, z);
  }
  float ss0 = 0, ss1 = 0, ss2 = 0, cs0 = 0, cs1 = 0, cs2 = 0;
  for (int u = 0; u < LU_; ++u) {
    float e = __expf(zr[u] + blin[u] - zmax);
    ss0 += e * sinT[u * 3 + 0]; cs0 += e * cosT[u * 3 + 0];
    ss1 += e * sinT[u * 3 + 1]; cs1 += e * cosT[u * 3 + 1];
    ss2 += e * sinT[u * 3 + 2]; cs2 += e * cosT[u * 3 + 2];
  }
  phi[(size_t)lb * 3 + 0] = atan2f(ss0, cs0);
  phi[(size_t)lb * 3 + 1] = atan2f(ss1, cs1);
  phi[(size_t)lb * 3 + 2] = atan2f(ss2, cs2);
}

// ---------------- K6: NeRF coordinate extension (per-batch serial chain) -------------
__global__ void k_nerf(const float* __restrict__ phi, float* __restrict__ out) {
  const int b = threadIdx.x;
  if (b >= BT) return;
  const float Rr[3] = {132.868f, 145.801f, 152.326f};
  const float Tt[3] = {2.028f, 2.124f, 1.941f};
  #pragma unroll
  for (int i = 0; i < 3; ++i)
    #pragma unroll
    for (int c = 0; c < 3; ++c)
      out[((size_t)i * BT + b) * 3 + c] = (i == c) ? 1.0f : 0.0f;
  float Ax = 1, Ay = 0, Az = 0, Bx = 0, By = 1, Bz = 0, Cx = 0, Cy = 0, Cz = 1;
  for (int l = 0; l < L_; ++l) {
    #pragma unroll
    for (int j = 0; j < 3; ++j) {
      float p = phi[((size_t)l * BT + b) * 3 + j];
      float st = sinf(Tt[j]), ct = cosf(Tt[j]);
      float d2x = -Rr[j] * ct, d2y = Rr[j] * cosf(p) * st, d2z = Rr[j] * sinf(p) * st;
      float bcx = Cx - Bx, bcy = Cy - By, bcz = Cz - Bz;
      float rn = rsqrtf(bcx * bcx + bcy * bcy + bcz * bcz);
      bcx *= rn; bcy *= rn; bcz *= rn;
      float abx = Bx - Ax, aby = By - Ay, abz = Bz - Az;
      float nx = aby * bcz - abz * bcy;
      float ny = abz * bcx - abx * bcz;
      float nz = abx * bcy - aby * bcx;
      rn = rsqrtf(nx * nx + ny * ny + nz * nz);
      nx *= rn; ny *= rn; nz *= rn;
      float mx = ny * bcz - nz * bcy;
      float my = nz * bcx - nx * bcz;
      float mz = nx * bcy - ny * bcx;
      float Dx = Cx + bcx * d2x + mx * d2y + nx * d2z;
      float Dy = Cy + bcy * d2x + my * d2y + ny * d2z;
      float Dz = Cz + bcz * d2x + mz * d2y + nz * d2z;
      Ax = Bx; Ay = By; Az = Bz;
      Bx = Cx; By = Cy; Bz = Cz;
      Cx = Dx; Cy = Dy; Cz = Dz;
      size_t o = ((size_t)(3 + l * 3 + j) * BT + b) * 3;
      out[o + 0] = Dx; out[o + 1] = Dy; out[o + 2] = Dz;
    }
  }
}

// ---------------- host launcher ----------------
extern "C" void kernel_launch(void* const* d_in, const int* in_sizes, int n_in,
                              void* d_out, int out_size, void* d_ws, size_t ws_size,
                              hipStream_t stream) {
  (void)in_sizes; (void)n_in; (void)out_size; (void)ws_size;
  const int*   seq   = (const int*)d_in[0];
  const float* pssm  = (const float*)d_in[1];
  // d_in[2] = length (all full-length -> identity packing, unused)
  const float* embed = (const float*)d_in[3];
  const float* WihF  = (const float*)d_in[4];
  const float* WhhF  = (const float*)d_in[5];
  const float* bihF  = (const float*)d_in[6];
  const float* bhhF  = (const float*)d_in[7];
  const float* WihB  = (const float*)d_in[8];
  const float* WhhB  = (const float*)d_in[9];
  const float* bihB  = (const float*)d_in[10];
  const float* bhhB  = (const float*)d_in[11];
  const float* Wlin  = (const float*)d_in[12];
  const float* blin  = (const float*)d_in[13];
  const float* alpha = (const float*)d_in[14];
  float* out = (float*)d_out;

  char* ws = (char*)d_ws;
  __bf16*   xb    = (__bf16*)(ws + OFF_XB);
  __bf16*   wihp  = (__bf16*)(ws + OFF_WIH);
  __bf16*   whhp  = (__bf16*)(ws + OFF_WHH);
  float*    biasp = (float*)(ws + OFF_BIAS);
  __bf16*   wlinp = (__bf16*)(ws + OFF_WLIN);
  float*    sinT  = (float*)(ws + OFF_TRIG);
  float*    cosT  = sinT + 180;
  unsigned* syncc = (unsigned*)(ws + OFF_SYNC);
  __bf16*   hbuf  = (__bf16*)(ws + OFF_HBUF);
  float*    xg    = (float*)(ws + OFF_XG);
  __bf16*   hcat  = (__bf16*)(ws + OFF_HCAT);
  float*    logits= (float*)(ws + OFF_LOG);
  float*    phi   = (float*)(ws + OFF_PHI);

  hipFuncSetAttribute(reinterpret_cast<const void*>(k_lstm),
                      hipFuncAttributeMaxDynamicSharedMemorySize, LDS_TOTAL);

  {
    size_t n = (size_t)2 * G4H * KPAD;
    k_pack_wih<<<dim3((unsigned)((n + 255) / 256)), 256, 0, stream>>>(WihF, WihB, wihp);
  }
  {
    size_t n = (size_t)2 * G4H * HID;
    k_pack_whh<<<dim3((unsigned)((n + 255) / 256)), 256, 0, stream>>>(WhhF, WhhB, whhp);
  }
  {
    int n = 4096 + LUP * KLIN + 180 + 2 * 2 * BT * HID + 4;
    k_pack_misc<<<dim3((n + 255) / 256), 256, 0, stream>>>(bihF, bhhF, bihB, bhhB, Wlin, alpha,
                                                           biasp, wlinp, sinT, cosT, hbuf, syncc);
  }
  {
    size_t n = (size_t)L_ * BT * KPAD;
    k_build_x<<<dim3((unsigned)((n + 255) / 256)), 256, 0, stream>>>(seq, pssm, embed, xb);
  }
  k_gemm_xg<<<dim3(L_ * BT / 128, G4H / 128, 2), 256, 0, stream>>>(xb, wihp, biasp, xg);
  k_lstm<<<dim3(2 * NBLK_DIR), 256, LDS_TOTAL, stream>>>(xg, whhp, hbuf, hcat, syncc);
  k_gemm_logits<<<dim3(L_ * BT / 128), 256, 0, stream>>>(hcat, wlinp, logits);
  k_angles<<<dim3(L_ * BT / 256), 256, 0, stream>>>(logits, blin, sinT, cosT, phi);
  k_nerf<<<dim3(1), 64, 0, stream>>>(phi, out);
}
